// AttentionLSTM_NoDropoutNoNorm_23012434772352
// MI455X (gfx1250) — matrix operand order repaired and checked by reading, not compiled, not run on hardware
//
#include <hip/hip_runtime.h>
#include <hip/hip_bf16.h>

// ---------------- problem constants ----------------
constexpr int BATCH = 4096;
constexpr int HID   = 300;          // hidden / input features
constexpr int TT    = 43;           // timesteps
constexpr int NCLS  = 80;
constexpr int GN    = 4 * HID;      // 1200 gate width
constexpr int KPAD  = 320;          // 300 padded to multiple of 32
constexpr int KTOT  = 2 * KPAD;     // 640 = [x | h] concat, padded
constexpr int KC    = 32;           // bf16 WMMA K per instruction
constexpr int NKC   = KTOT / KC;    // 20 k-chunks

// GEMM tiling: 256 threads = 8 waves; block tile 128(M) x 48(N)
constexpr int BM = 128;
constexpr int BN = 48;

typedef __bf16  v16bf __attribute__((ext_vector_type(16)));
typedef __bf16  v8bf  __attribute__((ext_vector_type(8)));
typedef float   v8f   __attribute__((ext_vector_type(8)));

__device__ __forceinline__ void async_lds_load32(unsigned lds_off, const void* gaddr) {
  // 32 bytes: two GLOBAL_LOAD_ASYNC_TO_LDS_B128; INST_OFFSET applies to both
  // the global and the LDS address (CDNA5 ISA).
  asm volatile("global_load_async_to_lds_b128 %0, %1, off"
               :: "v"(lds_off), "v"(gaddr) : "memory");
  asm volatile("global_load_async_to_lds_b128 %0, %1, off offset:16"
               :: "v"(lds_off), "v"(gaddr) : "memory");
}

__device__ __forceinline__ void wait_async0() {
#if __has_builtin(__builtin_amdgcn_s_wait_asynccnt)
  __builtin_amdgcn_s_wait_asynccnt(0);
#else
  asm volatile("s_wait_asynccnt 0" ::: "memory");
#endif
}

// ---------------- prep kernels ----------------

// x: [B][H][T] f32  ->  xbf: [T][B][KPAD] bf16 (zero padded j>=300)
__global__ __launch_bounds__(256) void k_xconvert(const float* __restrict__ x,
                                                  __bf16* __restrict__ xbf) {
  size_t idx = (size_t)blockIdx.x * 256 + threadIdx.x;
  size_t tot = (size_t)TT * BATCH * KPAD;
  if (idx >= tot) return;
  int t   = (int)(idx / ((size_t)BATCH * KPAD));
  int rem = (int)(idx - (size_t)t * BATCH * KPAD);
  int b = rem / KPAD;
  int j = rem - b * KPAD;
  float v = 0.f;
  if (j < HID) v = x[(size_t)b * HID * TT + (size_t)j * TT + t];
  xbf[idx] = (__bf16)v;
}

// Wt[d][n][k] = (k<300) w_ih[d][n][k] ; (320<=k<620) w_hh[d][n][k-320] ; else 0
// n-major so GEMM B-tiles are staged with contiguous 16B reads along k.
__global__ __launch_bounds__(256) void k_wpack(const float* __restrict__ w_ih,
                                               const float* __restrict__ w_hh,
                                               __bf16* __restrict__ Wt) {
  size_t idx = (size_t)blockIdx.x * 256 + threadIdx.x;
  size_t tot = (size_t)2 * GN * KTOT;
  if (idx >= tot) return;
  int d   = (int)(idx / ((size_t)GN * KTOT));
  int rem = (int)(idx - (size_t)d * GN * KTOT);
  int n = rem / KTOT;
  int k = rem - n * KTOT;
  float v = 0.f;
  if (k < HID)                          v = w_ih[(size_t)d * GN * HID + (size_t)n * HID + k];
  else if (k >= KPAD && k < KPAD + HID) v = w_hh[(size_t)d * GN * HID + (size_t)n * HID + (k - KPAD)];
  Wt[idx] = (__bf16)v;
}

// bias sum + zero-init c state and bf16 h state (padding lanes stay zero forever)
__global__ __launch_bounds__(256) void k_init(const float* __restrict__ b_ih,
                                              const float* __restrict__ b_hh,
                                              float* __restrict__ bsum,
                                              float* __restrict__ cst,
                                              __bf16* __restrict__ hbf) {
  size_t idx = (size_t)blockIdx.x * 256 + threadIdx.x;
  if (idx < (size_t)2 * GN)           bsum[idx] = b_ih[idx] + b_hh[idx];
  if (idx < (size_t)2 * BATCH * HID)  cst[idx] = 0.f;
  if (idx < (size_t)2 * BATCH * KPAD) hbf[idx] = (__bf16)0.f;
}

// ---------------- per-step gate GEMM (WMMA bf16, async-LDS staging) ----------------
// z[d][b][g] = sum_k A_d[b][k] * Wt[d][g][k],  A_d = [xbf[t][b][:] | hbf[d][b][:]]
__global__ __launch_bounds__(256) void k_gate_gemm(const __bf16* __restrict__ xbf,
                                                   const __bf16* __restrict__ hbf,
                                                   const __bf16* __restrict__ Wt,
                                                   float* __restrict__ z, int t) {
  const int d   = blockIdx.z;
  const int m0  = blockIdx.y * BM;
  const int n0  = blockIdx.x * BN;
  const int tid = threadIdx.x;
  const int wave = tid >> 5;
  const int lane = tid & 31;

  __shared__ __bf16 As[BM][KC] __attribute__((aligned(64)));  // [m][k]  8 KB
  __shared__ __bf16 Bs[BN][KC] __attribute__((aligned(64)));  // [n][k]  3 KB

  const __bf16* xrow = xbf + (size_t)t * BATCH * KPAD;
  const __bf16* hrow = hbf + (size_t)d * BATCH * KPAD;
  const __bf16* W    = Wt  + (size_t)d * GN * KTOT;

  v8f acc0 = {}; v8f acc1 = {}; v8f acc2 = {};

  const int arow = m0 + (tid >> 1);       // 128 rows, 2 threads/row
  const int aseg = (tid & 1) * 16;        // 16 bf16 (32 B) per thread

  const unsigned ldsA = (unsigned)(uintptr_t)&As[tid >> 1][aseg];
  const unsigned ldsB = (tid < 2 * BN)
      ? (unsigned)(uintptr_t)&Bs[tid >> 1][aseg] : 0u;
  const __bf16* srcBbase = W + (size_t)(n0 + (tid >> 1)) * KTOT + aseg;

  const int mrow  = wave * 16 + (lane & 15);  // A fragment row within block tile
  const int khalf = lane >> 4;                // 16-bit WMMA lane K-half

  for (int kc = 0; kc < NKC; ++kc) {
    const int kg = kc * KC;                   // 0..608, uniform
    // ---- stage A (128 x 32) direct to LDS, async ----
    const __bf16* srcA = (kg < KPAD)
        ? (xrow + (size_t)arow * KPAD + (kg + aseg))
        : (hrow + (size_t)arow * KPAD + (kg - KPAD + aseg));
    async_lds_load32(ldsA, srcA);
    if (kc + 1 < NKC) __builtin_prefetch(srcA + KC, 0, 1);   // global_prefetch
    // ---- stage B (48 x 32) direct to LDS, async (first 3 waves) ----
    if (tid < 2 * BN) async_lds_load32(ldsB, srcBbase + kg);
    wait_async0();
    __syncthreads();

    // ---- A fragment: lanes 0-15 hold K0..15, lanes 16-31 K16..31
    v8bf alo = *(const v8bf*)&As[mrow][khalf * 16];
    v8bf ahi = *(const v8bf*)&As[mrow][khalf * 16 + 8];
    v16bf afrag = __builtin_shufflevector(alo, ahi, 0,1,2,3,4,5,6,7,8,9,10,11,12,13,14,15);

    // ---- B fragments: lane<16 -> (n=lane, K0..15); lane>=16 -> (n=lane-16, K16..31)
    const int nn = lane & 15;
    v16bf b0 = *(const v16bf*)&Bs[nn][khalf * 16];
    v16bf b1 = *(const v16bf*)&Bs[16 + nn][khalf * 16];
    v16bf b2 = *(const v16bf*)&Bs[32 + nn][khalf * 16];

    acc0 = __builtin_amdgcn_wmma_f32_16x16x32_bf16(false, afrag, false, b0, (short)0, acc0, false, false);
    acc1 = __builtin_amdgcn_wmma_f32_16x16x32_bf16(false, afrag, false, b1, (short)0, acc1, false, false);
    acc2 = __builtin_amdgcn_wmma_f32_16x16x32_bf16(false, afrag, false, b2, (short)0, acc2, false, false);
    __syncthreads();
  }

  // ---- store C: vgpr r holds M = r + 8*khalf, N = lane&15
  float* zd = z + (size_t)d * BATCH * GN;
  const int crow = m0 + wave * 16 + khalf * 8;
  const int ccol = n0 + (lane & 15);
#pragma unroll
  for (int r = 0; r < 8; ++r) {
    zd[(size_t)(crow + r) * GN + ccol]       = acc0[r];
    zd[(size_t)(crow + r) * GN + ccol + 16]  = acc1[r];
    zd[(size_t)(crow + r) * GN + ccol + 32]  = acc2[r];
  }
}

// ---------------- per-step LSTM cell (elementwise, both dirs) ----------------
__global__ __launch_bounds__(256) void k_cell(const float* __restrict__ z,
                                              const float* __restrict__ bsum,
                                              float* __restrict__ cst,
                                              __bf16* __restrict__ hbf,
                                              float* __restrict__ hsum, int t) {
  int idx = blockIdx.x * 256 + threadIdx.x;
  if (idx >= BATCH * HID) return;
  int b = idx / HID;
  int j = idx - b * HID;
  float hs = 0.f;
#pragma unroll
  for (int d = 0; d < 2; ++d) {
    const float* zr = z + (size_t)d * BATCH * GN + (size_t)b * GN;
    const float* bb = bsum + d * GN;
    float ig = zr[j]           + bb[j];
    float fg = zr[HID + j]     + bb[HID + j];
    float gg = zr[2 * HID + j] + bb[2 * HID + j];
    float og = zr[3 * HID + j] + bb[3 * HID + j];
    float si = 1.f / (1.f + __expf(-ig));
    float sf = 1.f / (1.f + __expf(-fg));
    float so = 1.f / (1.f + __expf(-og));
    float tg = tanhf(gg);
    size_t ci = (size_t)d * BATCH * HID + idx;
    float c = sf * cst[ci] + si * tg;
    cst[ci] = c;
    float h = so * tanhf(c);
    hbf[(size_t)d * BATCH * KPAD + (size_t)b * KPAD + j] = (__bf16)h;
    hs += h;
  }
  hsum[(size_t)b * TT * HID + (size_t)t * HID + j] = hs;  // [B][T][H]
}

// ---------------- attention + FC + softmax (one block per batch row) ----------------
__global__ __launch_bounds__(256) void k_attn_fc(const float* __restrict__ hsum,
                                                 const float* __restrict__ conv_w,
                                                 const float* __restrict__ fc_w,
                                                 const float* __restrict__ fc_b,
                                                 float* __restrict__ out) {
  const int b = blockIdx.x;
  const int tid = threadIdx.x;
  const int lane = tid & 31, wave = tid >> 5;
  __shared__ float sc[TT];
  __shared__ float hstar[HID];
  __shared__ float logits[NCLS];
  const float* hb = hsum + (size_t)b * TT * HID;

  // attention scores: one wave per timestep
  for (int t = wave; t < TT; t += 8) {
    float p = 0.f;
    for (int j = lane; j < HID; j += 32) p += conv_w[j] * tanhf(hb[t * HID + j]);
    for (int off = 16; off; off >>= 1) p += __shfl_down(p, off, 32);
    if (lane == 0) sc[t] = p;
  }
  __syncthreads();
  // softmax over T (redundant per-thread reduce, then write)
  float m = -1e30f;
  for (int t = 0; t < TT; ++t) m = fmaxf(m, sc[t]);
  float s = 0.f;
  for (int t = 0; t < TT; ++t) s += __expf(sc[t] - m);
  float inv = 1.f / s;
  __syncthreads();
  if (tid < TT) sc[tid] = __expf(sc[tid] - m) * inv;
  __syncthreads();
  // r_j = sum_t h[t][j] * alpha[t] ; hStar = tanh(r)
  for (int j = tid; j < HID; j += 256) {
    float r = 0.f;
    for (int t = 0; t < TT; ++t) r += hb[t * HID + j] * sc[t];
    hstar[j] = tanhf(r);
  }
  __syncthreads();
  if (tid < NCLS) {
    float a = fc_b[tid];
    const float* wr = fc_w + tid * HID;
    for (int j = 0; j < HID; ++j) a += hstar[j] * wr[j];
    logits[tid] = a;
  }
  __syncthreads();
  float m2 = -1e30f;
  for (int c = 0; c < NCLS; ++c) m2 = fmaxf(m2, logits[c]);
  float s2 = 0.f;
  for (int c = 0; c < NCLS; ++c) s2 += __expf(logits[c] - m2);
  if (tid < NCLS) out[(size_t)b * NCLS + tid] = __expf(logits[tid] - m2) / s2;
}

// ---------------- host launcher ----------------
extern "C" void kernel_launch(void* const* d_in, const int* in_sizes, int n_in,
                              void* d_out, int out_size, void* d_ws, size_t ws_size,
                              hipStream_t stream) {
  const float* x      = (const float*)d_in[0];
  const float* w_ih   = (const float*)d_in[1];
  const float* w_hh   = (const float*)d_in[2];
  const float* b_ih   = (const float*)d_in[3];
  const float* b_hh   = (const float*)d_in[4];
  const float* conv_w = (const float*)d_in[5];
  const float* fc_w   = (const float*)d_in[6];
  const float* fc_b   = (const float*)d_in[7];
  float* out = (float*)d_out;

  char* ws = (char*)d_ws;
  auto carve = [&](size_t bytes) {
    char* p = ws;
    ws += (bytes + 255) & ~(size_t)255;
    return p;
  };
  __bf16* xbf  = (__bf16*)carve((size_t)TT * BATCH * KPAD * sizeof(__bf16));   // 112.7 MB
  __bf16* Wt   = (__bf16*)carve((size_t)2 * GN * KTOT * sizeof(__bf16));       //   3.1 MB
  float*  bsum = (float*) carve((size_t)2 * GN * sizeof(float));
  __bf16* hbf  = (__bf16*)carve((size_t)2 * BATCH * KPAD * sizeof(__bf16));    //   5.2 MB
  float*  cst  = (float*) carve((size_t)2 * BATCH * HID * sizeof(float));      //   9.8 MB
  float*  zbuf = (float*) carve((size_t)2 * BATCH * GN * sizeof(float));       //  39.3 MB
  float*  hsm  = (float*) carve((size_t)BATCH * TT * HID * sizeof(float));     // 211.4 MB
  (void)ws_size; (void)in_sizes; (void)n_in; (void)out_size;

  { size_t n = (size_t)TT * BATCH * KPAD;
    k_xconvert<<<(unsigned)((n + 255) / 256), 256, 0, stream>>>(x, xbf); }
  { size_t n = (size_t)2 * GN * KTOT;
    k_wpack<<<(unsigned)((n + 255) / 256), 256, 0, stream>>>(w_ih, w_hh, Wt); }
  { size_t n = (size_t)2 * BATCH * KPAD;
    k_init<<<(unsigned)((n + 255) / 256), 256, 0, stream>>>(b_ih, b_hh, bsum, cst, hbf); }

  dim3 gg(GN / BN, BATCH / BM, 2);   // (25, 32, 2)
  size_t ncell = (size_t)BATCH * HID;
  for (int t = 0; t < TT; ++t) {
    k_gate_gemm<<<gg, 256, 0, stream>>>(xbf, hbf, Wt, zbuf, t);
    k_cell<<<(unsigned)((ncell + 255) / 256), 256, 0, stream>>>(zbuf, bsum, cst, hbf, hsm, t);
  }
  k_attn_fc<<<BATCH, 256, 0, stream>>>(hsm, conv_w, fc_w, fc_b, out);
}
